// nAttention_18915035972298
// MI455X (gfx1250) — matrix operand order repaired and checked by reading, not compiled, not run on hardware
//
#include <hip/hip_runtime.h>

// ---------------------------------------------------------------------------
// nGPT-style attention for MI455X (gfx1250), bf16 WMMA everywhere.
// B=1, N=2048, E=1024, H=16, D=64.
// ---------------------------------------------------------------------------

#define SEQ   2048
#define EMB   1024
#define HEADS 16
#define HDIM  64

typedef __attribute__((ext_vector_type(16))) __bf16 v16bf;
typedef __attribute__((ext_vector_type(8)))  __bf16 v8bf;
typedef __attribute__((ext_vector_type(4)))  __bf16 v4bf;
typedef __attribute__((ext_vector_type(8)))  float  v8f;
typedef __attribute__((ext_vector_type(4)))  float  v4f;

union AV { v16bf v; v8bf h8[2]; };
union FV { v8f v; float f[8]; };

__device__ __forceinline__ __bf16 f2bf(float f) {
  unsigned u = __builtin_bit_cast(unsigned, f);
  unsigned r = (u + 0x7FFFu + ((u >> 16) & 1u)) >> 16;
  unsigned short s = (unsigned short)r;
  return __builtin_bit_cast(__bf16, s);
}

// ---------------------------------------------------------------------------
// Phase 1: f32 -> bf16, 4 elements / thread (b128 load, b64 store)
// ---------------------------------------------------------------------------
__global__ void cvt_bf16_kernel(const float* __restrict__ src,
                                __bf16* __restrict__ dst, int n4) {
  int i = blockIdx.x * blockDim.x + threadIdx.x;
  if (i < n4) {
    v4f x = *(const v4f*)(src + (size_t)i * 4);
    v4bf y;
    y[0] = f2bf(x[0]); y[1] = f2bf(x[1]);
    y[2] = f2bf(x[2]); y[3] = f2bf(x[3]);
    *(v4bf*)(dst + (size_t)i * 4) = y;
  }
}

// ---------------------------------------------------------------------------
// Phase 2/5: C[M,Nc] = A[M,K] * B[Nc,K]^T   (both operands K-contiguous)
// block = 128 threads (4 waves). Block tile 64x64; each wave: 16 rows x 64 cols.
// Named-register double-buffered K loop (no arrays, no runtime guards) so the
// 10 loads of chunk k+32 are in flight while the 4 WMMAs of chunk k issue.
// __launch_bounds__(128,1): lift the VGPR occupancy cap so buffers stay in
// registers (round 2 spilled to scratch/LDS under the default budget).
// Requires K % 64 == 0 (true here: K = 1024).
// ---------------------------------------------------------------------------
__global__ __launch_bounds__(128, 1)
void gemm_bf16_kernel(const __bf16* __restrict__ A,
                      const __bf16* __restrict__ Bw,
                      float* __restrict__ C, int M, int Nc, int K) {
  const int lane = threadIdx.x & 31;
  const int wave = threadIdx.x >> 5;
  const int tiles_n = Nc >> 6;
  const int bm = blockIdx.x / tiles_n;
  const int bn = blockIdx.x % tiles_n;
  const int m0 = bm * 64 + wave * 16;
  const int n0 = bn * 64;
  const int hf  = lane >> 4;   // 16-lane half
  const int l15 = lane & 15;

  // A fragment: lane row = m0+l15 ; halves own K-subranges {0..7,16..23}/{8..15,24..31}
  const __bf16* arow  = A + (size_t)(m0 + l15) * K + hf * 8;
  // B fragment: lane col = n0+nt*16+l15 ; halves own K-subranges {0..7,16..23}/{8..15,24..31}
  const __bf16* brow0 = Bw + (size_t)(n0 +  0 + l15) * K + hf * 8;
  const __bf16* brow1 = Bw + (size_t)(n0 + 16 + l15) * K + hf * 8;
  const __bf16* brow2 = Bw + (size_t)(n0 + 32 + l15) * K + hf * 8;
  const __bf16* brow3 = Bw + (size_t)(n0 + 48 + l15) * K + hf * 8;

  v8f acc0 = {}, acc1 = {}, acc2 = {}, acc3 = {};
  AV aA, bA0, bA1, bA2, bA3;   // buffer A (chunk in flight / being consumed)
  AV aB, bB0, bB1, bB2, bB3;   // buffer B

#define LOAD_CHUNK(aa, b0, b1, b2, b3, kk)                                    \
  do {                                                                        \
    aa.h8[0] = *(const v8bf*)(arow + (kk));                                   \
    aa.h8[1] = *(const v8bf*)(arow + (kk) + 16);                              \
    b0.h8[0] = *(const v8bf*)(brow0 + (kk));                                  \
    b0.h8[1] = *(const v8bf*)(brow0 + (kk) + 16);                             \
    b1.h8[0] = *(const v8bf*)(brow1 + (kk));                                  \
    b1.h8[1] = *(const v8bf*)(brow1 + (kk) + 16);                             \
    b2.h8[0] = *(const v8bf*)(brow2 + (kk));                                  \
    b2.h8[1] = *(const v8bf*)(brow2 + (kk) + 16);                             \
    b3.h8[0] = *(const v8bf*)(brow3 + (kk));                                  \
    b3.h8[1] = *(const v8bf*)(brow3 + (kk) + 16);                             \
  } while (0)

#define WMMA_CHUNK(aa, b0, b1, b2, b3)                                        \
  do {                                                                        \
    acc0 = __builtin_amdgcn_wmma_f32_16x16x32_bf16(false, aa.v, false, b0.v,  \
                                                   (short)0, acc0, false, false); \
    acc1 = __builtin_amdgcn_wmma_f32_16x16x32_bf16(false, aa.v, false, b1.v,  \
                                                   (short)0, acc1, false, false); \
    acc2 = __builtin_amdgcn_wmma_f32_16x16x32_bf16(false, aa.v, false, b2.v,  \
                                                   (short)0, acc2, false, false); \
    acc3 = __builtin_amdgcn_wmma_f32_16x16x32_bf16(false, aa.v, false, b3.v,  \
                                                   (short)0, acc3, false, false); \
  } while (0)

  // prologue: stage chunk 0 into buffer A
  LOAD_CHUNK(aA, bA0, bA1, bA2, bA3, 0);

  int k = 0;
  for (; k + 64 < K; k += 64) {
    LOAD_CHUNK(aB, bB0, bB1, bB2, bB3, k + 32);
    __builtin_prefetch(arow + k + 160, 0, 1);
    WMMA_CHUNK(aA, bA0, bA1, bA2, bA3);
    LOAD_CHUNK(aA, bA0, bA1, bA2, bA3, k + 64);
    WMMA_CHUNK(aB, bB0, bB1, bB2, bB3);
  }
  // tail: buffer A holds chunk K-64; fetch final chunk K-32 while consuming it
  LOAD_CHUNK(aB, bB0, bB1, bB2, bB3, K - 32);
  WMMA_CHUNK(aA, bA0, bA1, bA2, bA3);
  WMMA_CHUNK(aB, bB0, bB1, bB2, bB3);

#undef LOAD_CHUNK
#undef WMMA_CHUNK

  auto store_tile = [&](v8f av, int nt) {
    FV r; r.v = av;
    const int col = n0 + nt * 16 + l15;
#pragma unroll
    for (int j = 0; j < 8; ++j) {
      const int row = m0 + j + hf * 8;   // C layout: VGPR j -> row j (+8 hi half)
      C[(size_t)row * Nc + col] = r.f[j];
    }
  };
  store_tile(acc0, 0);
  store_tile(acc1, 1);
  store_tile(acc2, 2);
  store_tile(acc3, 3);
}

// ---------------------------------------------------------------------------
// Phase 3: RoPE + L2-normalize + sqk scale for q,k ; transpose v.
// One wave per (head, token). Lane l owns head-dims d=l and d=l+32 (the RoPE pair).
// ---------------------------------------------------------------------------
__global__ __launch_bounds__(256)
void rope_norm_kernel(const float* __restrict__ qf, const float* __restrict__ kf,
                      const float* __restrict__ vf, const float* __restrict__ sqk,
                      __bf16* __restrict__ qb, __bf16* __restrict__ kb,
                      __bf16* __restrict__ vtb) {
  const int lane = threadIdx.x & 31;
  const int wid  = blockIdx.x * (blockDim.x >> 5) + (threadIdx.x >> 5);
  const int n = wid & (SEQ - 1);
  const int h = wid >> 11;               // wid / SEQ
  if (h >= HEADS) return;
  const int e0 = h * HDIM;
  const size_t rowoff = (size_t)n * EMB + e0;

  // rope angle: inv_freq = 10000^(-(2*lane)/64)
  const float inv = __expf(-(2.0f * (float)lane / 64.0f) * 9.210340371976184f);
  const float fr  = (float)n * inv;
  float s, c;
  __sincosf(fr, &s, &c);

  const float sc1 = sqk[e0 + lane]       * 32.0f;   // sqk * sqrt(E)
  const float sc2 = sqk[e0 + 32 + lane]  * 32.0f;
  const size_t hb = (size_t)h * SEQ * HDIM + (size_t)n * HDIM;

#pragma unroll
  for (int which = 0; which < 2; ++which) {
    const float* src = which ? kf : qf;
    __bf16*      dst = which ? kb : qb;
    float t1 = src[rowoff + lane];
    float t2 = src[rowoff + 32 + lane];
    float r1 = t1 * c - t2 * s;
    float r2 = t1 * s + t2 * c;
    float ss = r1 * r1 + r2 * r2;
#pragma unroll
    for (int off = 16; off > 0; off >>= 1) ss += __shfl_xor(ss, off, 32);
    const float rn = rsqrtf(ss);
    dst[hb + lane]      = f2bf(r1 * rn * sc1);
    dst[hb + 32 + lane] = f2bf(r2 * rn * sc2);
  }

  // v: transpose to [H, D, N] so the PV WMMA B-operand is seq-contiguous
  float v1 = vf[rowoff + lane];
  float v2 = vf[rowoff + 32 + lane];
  vtb[((size_t)h * HDIM + lane)      * SEQ + n] = f2bf(v1);
  vtb[((size_t)h * HDIM + 32 + lane) * SEQ + n] = f2bf(v2);
}

// ---------------------------------------------------------------------------
// Phase 4: causal flash attention, one head+64-query-row block per workgroup.
// 4 waves; each wave owns 16 query rows, streams key blocks of 32.
// K fragments are batched before the S WMMAs; V fragments are issued before
// the softmax so their latency hides behind VALU + the LDS P round-trip.
// scores scale = sqrt(D) = 8 (as in the reference source).
// ---------------------------------------------------------------------------
__global__ __launch_bounds__(128, 1)
void flash_kernel(const __bf16* __restrict__ qb, const __bf16* __restrict__ kbm,
                  const __bf16* __restrict__ vtb, __bf16* __restrict__ aob) {
  __shared__ __align__(16) __bf16 lds_p[4][16 * 32];   // wave-private P tiles

  const int lane = threadIdx.x & 31;
  const int wave = threadIdx.x >> 5;
  const int h    = blockIdx.x & (HEADS - 1);
  const int qblk = blockIdx.x >> 4;
  const int qbase = qblk * 64 + wave * 16;
  const int hf  = lane >> 4;
  const int l15 = lane & 15;

  const __bf16* Q  = qb  + (size_t)h * SEQ * HDIM;
  const __bf16* Km = kbm + (size_t)h * SEQ * HDIM;
  const __bf16* Vt = vtb + (size_t)h * HDIM * SEQ;

  // Q A-fragments for d-chunks [0,32) and [32,64)
  const __bf16* qrow = Q + (size_t)(qbase + l15) * HDIM + hf * 8;
  AV qa0, qa1;
  qa0.h8[0] = *(const v8bf*)(qrow + 0);
  qa0.h8[1] = *(const v8bf*)(qrow + 16);
  qa1.h8[0] = *(const v8bf*)(qrow + 32);
  qa1.h8[1] = *(const v8bf*)(qrow + 48);

  // hoisted per-lane base pointers
  const __bf16* krow  = Km + (size_t)l15 * HDIM + hf * 8;      // + key*HDIM + dc*32
  const __bf16* vrow0 = Vt + (size_t)( 0 + l15) * SEQ + hf * 8;
  const __bf16* vrow1 = Vt + (size_t)(16 + l15) * SEQ + hf * 8;
  const __bf16* vrow2 = Vt + (size_t)(32 + l15) * SEQ + hf * 8;
  const __bf16* vrow3 = Vt + (size_t)(48 + l15) * SEQ + hf * 8;

  float mrow[8], lrow[8], alphaA[8];
#pragma unroll
  for (int j = 0; j < 8; ++j) { mrow[j] = -1e30f; lrow[j] = 0.0f; }
  v8f o0 = {}, o1 = {}, o2 = {}, o3 = {};

  __bf16* pt = &lds_p[wave][0];
  const int nkb = ((qbase + 15) >> 5) + 1;    // causal bound (key blocks of 32)

  for (int kb32 = 0; kb32 < nkb; ++kb32) {
    const int kpos0 = kb32 * 32;

    // ---- batch-load all 4 K fragments, then S = Q(16x64) @ K^T(64x32)
    AV bk00, bk01, bk10, bk11;   // [key-col tile][d-chunk]
    {
      const __bf16* kp0 = krow + (size_t)kpos0 * HDIM;
      const __bf16* kp1 = krow + (size_t)(kpos0 + 16) * HDIM;
      bk00.h8[0] = *(const v8bf*)(kp0);       bk00.h8[1] = *(const v8bf*)(kp0 + 16);
      bk01.h8[0] = *(const v8bf*)(kp0 + 32);  bk01.h8[1] = *(const v8bf*)(kp0 + 48);
      bk10.h8[0] = *(const v8bf*)(kp1);       bk10.h8[1] = *(const v8bf*)(kp1 + 16);
      bk11.h8[0] = *(const v8bf*)(kp1 + 32);  bk11.h8[1] = *(const v8bf*)(kp1 + 48);
    }
    __builtin_prefetch(krow + (size_t)(kpos0 + 32) * HDIM, 0, 1);

    v8f sacc0 = {}, sacc1 = {};
    sacc0 = __builtin_amdgcn_wmma_f32_16x16x32_bf16(false, qa0.v, false, bk00.v,
                                                    (short)0, sacc0, false, false);
    sacc0 = __builtin_amdgcn_wmma_f32_16x16x32_bf16(false, qa1.v, false, bk01.v,
                                                    (short)0, sacc0, false, false);
    sacc1 = __builtin_amdgcn_wmma_f32_16x16x32_bf16(false, qa0.v, false, bk10.v,
                                                    (short)0, sacc1, false, false);
    sacc1 = __builtin_amdgcn_wmma_f32_16x16x32_bf16(false, qa1.v, false, bk11.v,
                                                    (short)0, sacc1, false, false);

    // ---- issue V fragments now; latency hides behind softmax + LDS round-trip
    AV bv0, bv1, bv2, bv3;
    bv0.h8[0] = *(const v8bf*)(vrow0 + kpos0); bv0.h8[1] = *(const v8bf*)(vrow0 + kpos0 + 16);
    bv1.h8[0] = *(const v8bf*)(vrow1 + kpos0); bv1.h8[1] = *(const v8bf*)(vrow1 + kpos0 + 16);
    bv2.h8[0] = *(const v8bf*)(vrow2 + kpos0); bv2.h8[1] = *(const v8bf*)(vrow2 + kpos0 + 16);
    bv3.h8[0] = *(const v8bf*)(vrow3 + kpos0); bv3.h8[1] = *(const v8bf*)(vrow3 + kpos0 + 16);

    // ---- scale, causal mask, online softmax (rows live in 16-lane halves)
    FV s0, s1; s0.v = sacc0; s1.v = sacc1;
#pragma unroll
    for (int j = 0; j < 8; ++j) {
      const int row = qbase + j + hf * 8;
      float a0 = (kpos0 + l15      <= row) ? s0.f[j] * 8.0f : -1e30f;
      float a1 = (kpos0 + 16 + l15 <= row) ? s1.f[j] * 8.0f : -1e30f;
      float m = fmaxf(a0, a1);
      m = fmaxf(m, __shfl_xor(m, 1, 32));
      m = fmaxf(m, __shfl_xor(m, 2, 32));
      m = fmaxf(m, __shfl_xor(m, 4, 32));
      m = fmaxf(m, __shfl_xor(m, 8, 32));
      const float mnew = fmaxf(mrow[j], m);
      const float alpha = __expf(mrow[j] - mnew);
      mrow[j] = mnew;
      alphaA[j] = alpha;
      float p0 = __expf(a0 - mnew);
      float p1 = __expf(a1 - mnew);
      s0.f[j] = p0; s1.f[j] = p1;
      float ps = p0 + p1;
      ps += __shfl_xor(ps, 1, 32);
      ps += __shfl_xor(ps, 2, 32);
      ps += __shfl_xor(ps, 4, 32);
      ps += __shfl_xor(ps, 8, 32);
      lrow[j] = lrow[j] * alpha + ps;
    }
    {
      FV t0, t1, t2, t3;
      t0.v = o0; t1.v = o1; t2.v = o2; t3.v = o3;
#pragma unroll
      for (int j = 0; j < 8; ++j) {
        t0.f[j] *= alphaA[j]; t1.f[j] *= alphaA[j];
        t2.f[j] *= alphaA[j]; t3.f[j] *= alphaA[j];
      }
      o0 = t0.v; o1 = t1.v; o2 = t2.v; o3 = t3.v;
    }

    // ---- P: C-layout -> LDS -> A-layout (wave-private; in-order LDS + wait)
#pragma unroll
    for (int j = 0; j < 8; ++j) {
      pt[(j + hf * 8) * 32 + l15]      = f2bf(s0.f[j]);
      pt[(j + hf * 8) * 32 + 16 + l15] = f2bf(s1.f[j]);
    }
    asm volatile("s_wait_dscnt 0" ::: "memory");
    AV pa;
    pa.h8[0] = *(const v8bf*)(pt + l15 * 32 + hf * 8);
    pa.h8[1] = *(const v8bf*)(pt + l15 * 32 + hf * 8 + 16);

    // ---- O += P(16x32) @ V(32x64): B from transposed V (seq-contiguous)
    o0 = __builtin_amdgcn_wmma_f32_16x16x32_bf16(false, pa.v, false, bv0.v,
                                                 (short)0, o0, false, false);
    o1 = __builtin_amdgcn_wmma_f32_16x16x32_bf16(false, pa.v, false, bv1.v,
                                                 (short)0, o1, false, false);
    o2 = __builtin_amdgcn_wmma_f32_16x16x32_bf16(false, pa.v, false, bv2.v,
                                                 (short)0, o2, false, false);
    o3 = __builtin_amdgcn_wmma_f32_16x16x32_bf16(false, pa.v, false, bv3.v,
                                                 (short)0, o3, false, false);
  }

  // ---- normalize and emit in token-major [N, E] bf16 for the Wo GEMM
  auto emit_tile = [&](v8f ov, int nt) {
    FV r; r.v = ov;
#pragma unroll
    for (int j = 0; j < 8; ++j) {
      const int row = qbase + j + hf * 8;
      aob[(size_t)row * EMB + h * HDIM + nt * 16 + l15] =
          f2bf(r.f[j] / lrow[j]);
    }
  };
  emit_tile(o0, 0);
  emit_tile(o1, 1);
  emit_tile(o2, 2);
  emit_tile(o3, 3);
}

// ---------------------------------------------------------------------------
// Host-side launch (graph-capture safe: only kernel launches on `stream`).
// Workspace layout (52 MB total):
//   [0,4M) x_bf16 | [4,12M) Wq/Wk/Wv/Wo bf16 | [12,36M) q/k/v f32
//   [36,44M) q/k bf16 [H,N,D] | [44,48M) v^T bf16 [H,D,N] | [48,52M) attn-out bf16
// ---------------------------------------------------------------------------
extern "C" void kernel_launch(void* const* d_in, const int* in_sizes, int n_in,
                              void* d_out, int out_size, void* d_ws, size_t ws_size,
                              hipStream_t stream) {
  const float* x   = (const float*)d_in[0];
  const float* Wq  = (const float*)d_in[1];
  const float* Wk  = (const float*)d_in[2];
  const float* Wv  = (const float*)d_in[3];
  const float* Wo  = (const float*)d_in[4];
  const float* sqk = (const float*)d_in[5];
  float* out = (float*)d_out;

  char* ws = (char*)d_ws;
  constexpr size_t MB = 1u << 20;
  __bf16* xb  = (__bf16*)(ws + 0);
  __bf16* wqb = (__bf16*)(ws + 4 * MB);
  __bf16* wkb = (__bf16*)(ws + 6 * MB);
  __bf16* wvb = (__bf16*)(ws + 8 * MB);
  __bf16* wob = (__bf16*)(ws + 10 * MB);
  float*  qf  = (float*)(ws + 12 * MB);
  float*  kf  = (float*)(ws + 20 * MB);
  float*  vf  = (float*)(ws + 28 * MB);
  __bf16* qb  = (__bf16*)(ws + 36 * MB);
  __bf16* kb  = (__bf16*)(ws + 40 * MB);
  __bf16* vtb = (__bf16*)(ws + 44 * MB);
  __bf16* aob = (__bf16*)(ws + 48 * MB);

  const int nX4 = SEQ * EMB / 4;     // 512 K vec4
  const int nW4 = EMB * EMB / 4;     // 256 K vec4

  // Phase 1: convert to bf16
  cvt_bf16_kernel<<<(nX4 + 255) / 256, 256, 0, stream>>>(x, xb, nX4);
  cvt_bf16_kernel<<<(nW4 + 255) / 256, 256, 0, stream>>>(Wq, wqb, nW4);
  cvt_bf16_kernel<<<(nW4 + 255) / 256, 256, 0, stream>>>(Wk, wkb, nW4);
  cvt_bf16_kernel<<<(nW4 + 255) / 256, 256, 0, stream>>>(Wv, wvb, nW4);
  cvt_bf16_kernel<<<(nW4 + 255) / 256, 256, 0, stream>>>(Wo, wob, nW4);

  // Phase 2: q/k/v projections  (grid = (2048/64)*(1024/64) = 512 blocks)
  const int gemm_blocks = (SEQ / 64) * (EMB / 64);
  gemm_bf16_kernel<<<gemm_blocks, 128, 0, stream>>>(xb, wqb, qf, SEQ, EMB, EMB);
  gemm_bf16_kernel<<<gemm_blocks, 128, 0, stream>>>(xb, wkb, kf, SEQ, EMB, EMB);
  gemm_bf16_kernel<<<gemm_blocks, 128, 0, stream>>>(xb, wvb, vf, SEQ, EMB, EMB);

  // Phase 3: RoPE + norm + scale; v transpose.  16*2048 waves, 8 waves/block
  rope_norm_kernel<<<(HEADS * SEQ) / 8, 256, 0, stream>>>(qf, kf, vf, sqk,
                                                          qb, kb, vtb);

  // Phase 4: causal flash attention. 16 heads * 32 q-blocks
  flash_kernel<<<HEADS * (SEQ / 64), 128, 0, stream>>>(qb, kb, vtb, aob);

  // Phase 5: output projection straight into d_out (f32)
  gemm_bf16_kernel<<<gemm_blocks, 128, 0, stream>>>(aob, wob, out, SEQ, EMB, EMB);
}
